// MultiHeadAttention_40853728920141
// MI455X (gfx1250) — matrix operand order repaired and checked by reading, not compiled, not run on hardware
//
#include <hip/hip_runtime.h>
#include <hip/hip_bf16.h>

// MI455X / gfx1250 multi-head attention forward.
// All matmuls use v_wmma_f32_16x16x32_bf16 (bf16 A/B, f32 accumulate).
// Attention is computed transposed (S^T = K Q^T, O^T = V^T P^T) so the
// softmax key-reduction is in-lane over C-fragment registers (keys = rows)
// plus one shfl_xor(16) to merge lane halves; running max/sum are per-lane
// scalars and the O rescale is a plain per-lane multiply.
// Conversion-free LDS staging (K tile, bf16 GEMM A tile) uses
// GLOBAL_LOAD_ASYNC_TO_LDS_B128 when the toolchain exposes the builtin.
//
// Workspace layout (32 MB total):
//   q  bf16 [B*H][S][hd]   8 MB  (offset 0)   (pre-scaled by hd^-0.5 in RoPE)
//   k  bf16 [B*H][S][hd]   8 MB  (offset 8 MB)
//   v  bf16 [B*H][S][hd]   8 MB  (offset 16 MB)
//   ao bf16 [B*S][D]       8 MB  (offset 24 MB)

typedef __attribute__((ext_vector_type(16))) __bf16 v16bf;
typedef __attribute__((ext_vector_type(8)))  __bf16 v8bf;
typedef __attribute__((ext_vector_type(8)))  float  v8f;
typedef __attribute__((ext_vector_type(4)))  int    v4i;

#define CB  2       // batch
#define CS  2048    // seq
#define CD  1024    // model dim
#define CH  16      // heads
#define CHD 64      // head dim
#define PER_T ((size_t)CB * CH * CS * CHD)   // 4M elements per q/k/v tensor

#if __has_builtin(__builtin_amdgcn_global_load_async_to_lds_b128) && \
    __has_builtin(__builtin_amdgcn_s_wait_asynccnt)
#define USE_ASYNC_LDS 1
#define AS1 __attribute__((address_space(1)))
#define AS3 __attribute__((address_space(3)))
// global -> LDS b128 async copy (per-lane addresses), tracked by ASYNCcnt.
// Builtin signature on this toolchain: (AS1 v4i* src, AS3 v4i* dst, Ii, Ii).
__device__ __forceinline__ void async_copy_b128(const void* gsrc, void* ldst) {
    __builtin_amdgcn_global_load_async_to_lds_b128(
        (AS1 v4i*)(unsigned long long)(uintptr_t)gsrc,   // global VA == generic VA
        (AS3 v4i*)ldst,                                  // addrspacecast to LDS
        0, 0);
}
__device__ __forceinline__ void async_wait0() {
    __builtin_amdgcn_s_wait_asynccnt(0);
}
#else
#define USE_ASYNC_LDS 0
#endif

// ---------------------------------------------------------------------------
// Generic bf16-WMMA GEMM: C = A(MxK) * W(KxN) + bias
//   SRC_F32: A source is f32 (converted to bf16 while staging) else bf16
//            (bf16 path uses async global->LDS when available).
//   MODE 0 : scatter bf16 into qkv [3][B][H][S][hd] layout (QKV projection)
//   MODE 1 : write f32 row-major MxN (output projection)
// Block tile 128x128, BK=32; 8 waves arranged 2(m) x 4(n), each wave 64x32
// = 4x2 WMMA tiles; 8 v_wmma per wave per k-step.
// ---------------------------------------------------------------------------
template<bool SRC_F32, int MODE>
__global__ __launch_bounds__(256)
void gemm_bf16_wmma(const void* __restrict__ Asrc,
                    const float* __restrict__ W,
                    const float* __restrict__ bias,
                    __bf16* __restrict__ qkv_out,
                    float*  __restrict__ f32_out,
                    int M, int N, int K)
{
    constexpr int BM = 128, BN = 128, BK = 32;
    __shared__ __attribute__((aligned(32))) __bf16 lA[BM * BK];     // [m][k]
    __shared__ __attribute__((aligned(32))) __bf16 lB[BN * BK];     // [n][k] (transposed)

    const int tid  = threadIdx.x;
    const int wave = tid >> 5;
    const int lane = tid & 31;
    const int hlf  = lane >> 4;     // lane-group (0: lanes 0-15, 1: lanes 16-31)
    const int l16  = lane & 15;
    const int wm   = (wave >> 2) * 64;   // wave m-offset within block tile
    const int wn   = (wave & 3) * 32;    // wave n-offset within block tile
    const int bm   = blockIdx.x * BM;
    const int bn   = blockIdx.y * BN;

    v8f acc[4][2];
#pragma unroll
    for (int mt = 0; mt < 4; ++mt)
#pragma unroll
        for (int nt = 0; nt < 2; ++nt)
#pragma unroll
            for (int r = 0; r < 8; ++r) acc[mt][nt][r] = 0.0f;

    for (int k0 = 0; k0 < K; k0 += BK) {
        // ---- stage A tile (BM x BK) into LDS as bf16, row-major ----
        if constexpr (SRC_F32) {
            const float* A = (const float*)Asrc;
#pragma unroll
            for (int i = 0; i < 4; ++i) {
                int e   = tid + i * 256;          // 1024 float4 chunks
                int row = e >> 3;
                int kc  = (e & 7) << 2;
                float4 f = *reinterpret_cast<const float4*>(
                    A + (size_t)(bm + row) * K + k0 + kc);
                union { __bf16 b[4]; uint2 u; } pk;
                pk.b[0] = (__bf16)f.x; pk.b[1] = (__bf16)f.y;
                pk.b[2] = (__bf16)f.z; pk.b[3] = (__bf16)f.w;
                *reinterpret_cast<uint2*>(&lA[row * BK + kc]) = pk.u;
            }
        } else {
            const __bf16* A = (const __bf16*)Asrc;
#pragma unroll
            for (int i = 0; i < 2; ++i) {
                int e   = tid + i * 256;          // 512 v8bf (16B) chunks
                int row = e >> 2;
                int kc  = (e & 3) << 3;
                const __bf16* gp = A + (size_t)(bm + row) * K + k0 + kc;
                __bf16* lp = &lA[row * BK + kc];
#if USE_ASYNC_LDS
                async_copy_b128(gp, lp);          // global_load_async_to_lds_b128
#else
                *reinterpret_cast<v8bf*>(lp) = *reinterpret_cast<const v8bf*>(gp);
#endif
            }
        }
        // ---- stage W tile (BK x BN) transposed into LDS [n][k] as bf16 ----
#pragma unroll
        for (int i = 0; i < 4; ++i) {
            int e  = tid + i * 256;               // 1024 float4 chunks
            int kr = e >> 5;                      // 0..31
            int nc = (e & 31) << 2;               // 0..124
            float4 f = *reinterpret_cast<const float4*>(
                W + (size_t)(k0 + kr) * N + bn + nc);
            lB[(nc + 0) * BK + kr] = (__bf16)f.x;
            lB[(nc + 1) * BK + kr] = (__bf16)f.y;
            lB[(nc + 2) * BK + kr] = (__bf16)f.z;
            lB[(nc + 3) * BK + kr] = (__bf16)f.w;
        }
#if USE_ASYNC_LDS
        if constexpr (!SRC_F32) async_wait0();    // drain ASYNCcnt before barrier
#endif
        __syncthreads();

        // ---- fragments + WMMA ----
        v16bf af[4], bfrag[2];
#pragma unroll
        for (int mt = 0; mt < 4; ++mt) {
            const __bf16* rp = &lA[(wm + mt * 16 + l16) * BK];
            v8bf lo = *reinterpret_cast<const v8bf*>(rp + hlf * 16);       // k = hlf*16 + 0..7
            v8bf hi = *reinterpret_cast<const v8bf*>(rp + hlf * 16 + 8);   // k = hlf*16 + 8..15
#pragma unroll
            for (int i = 0; i < 8; ++i) { af[mt][i] = lo[i]; af[mt][i + 8] = hi[i]; }
        }
#pragma unroll
        for (int nt = 0; nt < 2; ++nt)
            bfrag[nt] = *reinterpret_cast<const v16bf*>(
                &lB[(wn + nt * 16 + l16) * BK + hlf * 16]);                // k = hlf*16 + i

#pragma unroll
        for (int mt = 0; mt < 4; ++mt)
#pragma unroll
            for (int nt = 0; nt < 2; ++nt)
                acc[mt][nt] = __builtin_amdgcn_wmma_f32_16x16x32_bf16(
                    false, af[mt], false, bfrag[nt], (short)0, acc[mt][nt], false, false);
        __syncthreads();
    }

    // ---- epilogue ----
#pragma unroll
    for (int mt = 0; mt < 4; ++mt)
#pragma unroll
        for (int nt = 0; nt < 2; ++nt)
#pragma unroll
            for (int r = 0; r < 8; ++r) {
                int m = bm + wm + mt * 16 + r + hlf * 8;
                int n = bn + wn + nt * 16 + l16;
                float vv = acc[mt][nt][r] + bias[n];
                if constexpr (MODE == 0) {
                    int which = n >> 10, rest = n & 1023;
                    int h = rest >> 6, d = rest & 63;
                    int bb = m >> 11, s = m & 2047;
                    qkv_out[(size_t)which * PER_T +
                            (((size_t)(bb * CH + h) * CS + s) * CHD + d)] = (__bf16)vv;
                } else {
                    f32_out[(size_t)m * N + n] = vv;
                }
            }
}

// ---------------------------------------------------------------------------
// RoPE applied in place on bf16 q and k, [B*H][S][hd] layout.
// The attention scale hd^-0.5 is folded into q here (free), so the flash
// kernel consumes raw dot products.
// ---------------------------------------------------------------------------
__global__ __launch_bounds__(256)
void rope_kernel(__bf16* __restrict__ q, __bf16* __restrict__ k,
                 const float* __restrict__ cs, const float* __restrict__ sn)
{
    const float scale = 0.125f;                    // 64^-0.5
    int t = blockIdx.x * 256 + threadIdx.x;        // over B*H*S*(hd/2) = 2M pairs
    int j = t & 31;
    int s = (t >> 5) & (CS - 1);
    size_t idx = ((size_t)(t >> 5)) * CHD + 2 * j;
    float c  = cs[s * 32 + j];
    float si = sn[s * 32 + j];
    float qe = (float)q[idx], qo = (float)q[idx + 1];
    q[idx]     = (__bf16)((qe * c - qo * si) * scale);
    q[idx + 1] = (__bf16)((qe * si + qo * c) * scale);
    float ke = (float)k[idx], ko = (float)k[idx + 1];
    k[idx]     = (__bf16)(ke * c - ko * si);
    k[idx + 1] = (__bf16)(ke * si + ko * c);
}

// ---------------------------------------------------------------------------
// Flash attention, causal, TRANSPOSED orientation.
// Block = 256 threads = 8 waves over one (b,h); wave owns 32 queries x hd=64.
//   S^T = K · Q^T  (rows = keys, cols = queries)
//   O^T = V^T · P^T (rows = hd,  cols = queries)
// Per-lane column ownership (q = lane&15) => softmax stats are 2 scalars,
// key-reduction is in-lane + one shfl_xor(16); O rescale is per-lane mul.
// ---------------------------------------------------------------------------
__global__ __launch_bounds__(256)
void flash_attn_kernel(const __bf16* __restrict__ Q,
                       const __bf16* __restrict__ Kt,
                       const __bf16* __restrict__ V,
                       __bf16* __restrict__ Aout)
{
    __shared__ __attribute__((aligned(32))) __bf16 sK [32 * 64];    // [key][hd]
    __shared__ __attribute__((aligned(32))) __bf16 sVt[64 * 32];    // [hd][key]
    __shared__ __attribute__((aligned(32))) __bf16 sP [8][32 * 32]; // per-wave P^T as [q][key]

    const int tid  = threadIdx.x;
    const int wave = tid >> 5;
    const int lane = tid & 31;
    const int hlf  = lane >> 4;
    const int l16  = lane & 15;
    const int bh   = blockIdx.y;                 // 0..B*H-1
    const int qblk = blockIdx.x;                 // 0..S/256-1
    const int qw   = qblk * 256 + wave * 32;     // wave's first query row
    const size_t base = (size_t)bh * CS * CHD;

    // Q as B-fragments of S^T: b[i] = Q[q][hd = kk*32 + hlf*16 + i]
    v16bf qb[2][2];
#pragma unroll
    for (int nt = 0; nt < 2; ++nt) {
        const __bf16* rp = Q + base + (size_t)(qw + nt * 16 + l16) * CHD;
#pragma unroll
        for (int kk = 0; kk < 2; ++kk)
            qb[nt][kk] = *reinterpret_cast<const v16bf*>(rp + kk * 32 + hlf * 16);
    }

    v8f o[4][2];                      // O^T: [d-tile 0..3][q-tile 0..1]
#pragma unroll
    for (int mt = 0; mt < 4; ++mt)
#pragma unroll
        for (int nt = 0; nt < 2; ++nt)
#pragma unroll
            for (int r = 0; r < 8; ++r) o[mt][nt][r] = 0.0f;
    float mrun[2] = { -1e30f, -1e30f };
    float lrun[2] = { 0.0f, 0.0f };

    const int nkt = (qblk + 1) * 8;   // causal: 32-wide key tiles
    for (int kt = 0; kt < nkt; ++kt) {
        const int kb = kt * 32;
        // ---- stage K (row-major, async when available) and V (transposed);
        //      prefetch next tile ----
        {
            int key = tid >> 3, c8 = (tid & 7) << 3;
            const __bf16* kp = Kt + base + (size_t)(kb + key) * CHD + c8;
            const __bf16* vp = V  + base + (size_t)(kb + key) * CHD + c8;
#if USE_ASYNC_LDS
            async_copy_b128(kp, &sK[key * 64 + c8]);   // global_load_async_to_lds_b128
#else
            *reinterpret_cast<v8bf*>(&sK[key * 64 + c8]) =
                *reinterpret_cast<const v8bf*>(kp);
#endif
            v8bf vv = *reinterpret_cast<const v8bf*>(vp);
#pragma unroll
            for (int j = 0; j < 8; ++j) sVt[(c8 + j) * 32 + key] = vv[j];
            if (kt + 1 < nkt) {                 // global_prefetch_b8 next K/V tile
                __builtin_prefetch(kp + 32 * CHD, 0, 0);
                __builtin_prefetch(vp + 32 * CHD, 0, 0);
            }
        }
#if USE_ASYNC_LDS
        async_wait0();                          // drain ASYNCcnt before barrier
#endif
        __syncthreads();

        if (kb <= qw + 31) {          // wave-uniform: EXEC stays all-ones
            // ---- S^T = K Q^T : st[key-tile][q-tile] ----
            v8f st[2][2];
#pragma unroll
            for (int mt = 0; mt < 2; ++mt)
#pragma unroll
                for (int nt = 0; nt < 2; ++nt)
#pragma unroll
                    for (int r = 0; r < 8; ++r) st[mt][nt][r] = 0.0f;
#pragma unroll
            for (int kk = 0; kk < 2; ++kk) {
                v16bf ak[2];
#pragma unroll
                for (int mt = 0; mt < 2; ++mt) {
                    const __bf16* rp = &sK[(mt * 16 + l16) * 64 + kk * 32];
                    v8bf lo = *reinterpret_cast<const v8bf*>(rp + hlf * 16);
                    v8bf hi = *reinterpret_cast<const v8bf*>(rp + hlf * 16 + 8);
#pragma unroll
                    for (int i = 0; i < 8; ++i) { ak[mt][i] = lo[i]; ak[mt][i + 8] = hi[i]; }
                }
#pragma unroll
                for (int mt = 0; mt < 2; ++mt)
#pragma unroll
                    for (int nt = 0; nt < 2; ++nt)
                        st[mt][nt] = __builtin_amdgcn_wmma_f32_16x16x32_bf16(
                            false, ak[mt], false, qb[nt][kk], (short)0,
                            st[mt][nt], false, false);
            }

            // ---- causal mask: only diagonal tiles pay for it ----
            if (kb + 31 > qw) {
#pragma unroll
                for (int mt = 0; mt < 2; ++mt)
#pragma unroll
                    for (int r = 0; r < 8; ++r) {
                        int key = kb + mt * 16 + r + hlf * 8;
#pragma unroll
                        for (int nt = 0; nt < 2; ++nt) {
                            int qq = qw + nt * 16 + l16;
                            if (key > qq) st[mt][nt][r] = -1e9f;
                        }
                    }
            }

            // ---- online softmax per query column (per lane) ----
#pragma unroll
            for (int nt = 0; nt < 2; ++nt) {
                float rmax = st[0][nt][0];
#pragma unroll
                for (int mt = 0; mt < 2; ++mt)
#pragma unroll
                    for (int r = 0; r < 8; ++r)
                        rmax = fmaxf(rmax, st[mt][nt][r]);
                rmax = fmaxf(rmax, __shfl_xor(rmax, 16, 32));   // merge lane halves
                float mnew = fmaxf(mrun[nt], rmax);
                float corr = __expf(mrun[nt] - mnew);
                mrun[nt] = mnew;
                float rsum = 0.0f;
#pragma unroll
                for (int mt = 0; mt < 2; ++mt) {
                    v8bf pk;
#pragma unroll
                    for (int r = 0; r < 8; ++r) {
                        float p = __expf(st[mt][nt][r] - mnew);
                        rsum += p;
                        pk[r] = (__bf16)p;
                    }
                    // P^T stored [q][key]; 8 consecutive keys -> one b128 store
                    *reinterpret_cast<v8bf*>(
                        &sP[wave][(nt * 16 + l16) * 32 + mt * 16 + hlf * 8]) = pk;
                }
                rsum += __shfl_xor(rsum, 16, 32);
                lrun[nt] = lrun[nt] * corr + rsum;
#pragma unroll
                for (int mt = 0; mt < 4; ++mt)
#pragma unroll
                    for (int r = 0; r < 8; ++r) o[mt][nt][r] *= corr;
            }

            // ---- O^T += V^T P^T ----
            v16bf pb[2];
#pragma unroll
            for (int nt = 0; nt < 2; ++nt)     // b[i] = P^T[key = i+hlf*16][q]
                pb[nt] = *reinterpret_cast<const v16bf*>(
                    &sP[wave][(nt * 16 + l16) * 32 + hlf * 16]);
#pragma unroll
            for (int mt = 0; mt < 4; ++mt) {
                const __bf16* rp = &sVt[(mt * 16 + l16) * 32];
                v8bf lo = *reinterpret_cast<const v8bf*>(rp + hlf * 16);
                v8bf hi = *reinterpret_cast<const v8bf*>(rp + hlf * 16 + 8);
                v16bf av;
#pragma unroll
                for (int i = 0; i < 8; ++i) { av[i] = lo[i]; av[i + 8] = hi[i]; }
#pragma unroll
                for (int nt = 0; nt < 2; ++nt)
                    o[mt][nt] = __builtin_amdgcn_wmma_f32_16x16x32_bf16(
                        false, av, false, pb[nt], (short)0, o[mt][nt], false, false);
            }
        }
        __syncthreads();
    }

    // ---- epilogue: O^T / l, re-stage via per-wave LDS so each lane writes
    //      64 contiguous bytes of Aout[b][s][h*hd + d] ----
    const int bb = bh >> 4, h = bh & 15;
    float inv[2] = { 1.0f / lrun[0], 1.0f / lrun[1] };
#pragma unroll
    for (int dh = 0; dh < 2; ++dh) {
#pragma unroll
        for (int mtl = 0; mtl < 2; ++mtl) {
            int mt = dh * 2 + mtl;
#pragma unroll
            for (int nt = 0; nt < 2; ++nt) {
                v8bf pk;
#pragma unroll
                for (int r = 0; r < 8; ++r)
                    pk[r] = (__bf16)(o[mt][nt][r] * inv[nt]);
                // sP as [q_local][d_local]
                *reinterpret_cast<v8bf*>(
                    &sP[wave][(nt * 16 + l16) * 32 + mtl * 16 + hlf * 8]) = pk;
            }
        }
        const __bf16* rp = &sP[wave][lane * 32];
        v16bf w0 = *reinterpret_cast<const v16bf*>(rp);
        v16bf w1 = *reinterpret_cast<const v16bf*>(rp + 16);
        __bf16* dst = Aout + ((size_t)bb * CS + qw + lane) * CD + h * CHD + dh * 32;
        *reinterpret_cast<v16bf*>(dst)      = w0;
        *reinterpret_cast<v16bf*>(dst + 16) = w1;
    }
}

// ---------------------------------------------------------------------------
extern "C" void kernel_launch(void* const* d_in, const int* in_sizes, int n_in,
                              void* d_out, int out_size, void* d_ws, size_t ws_size,
                              hipStream_t stream)
{
    const float* x     = (const float*)d_in[0];
    const float* w_qkv = (const float*)d_in[1];
    const float* b_qkv = (const float*)d_in[2];
    const float* w_out = (const float*)d_in[3];
    const float* b_out = (const float*)d_in[4];
    const float* cos_t = (const float*)d_in[5];
    const float* sin_t = (const float*)d_in[6];
    // d_in[7] (mask) is implicit: causal tril handled analytically.

    __bf16* q    = (__bf16*)d_ws;
    __bf16* k    = q + PER_T;
    __bf16* v    = k + PER_T;
    __bf16* attn = v + PER_T;     // total 32 MB workspace

    // 1) QKV projection: (4096 x 1024) @ (1024 x 3072) -> bf16 qkv scatter
    {
        dim3 grid(4096 / 128, 3072 / 128);
        gemm_bf16_wmma<true, 0><<<grid, 256, 0, stream>>>(
            x, w_qkv, b_qkv, q, nullptr, CB * CS, 3 * CD, CD);
    }
    // 2) RoPE on q, k (+ fold attention scale into q)
    {
        int pairs = CB * CH * CS * (CHD / 2);    // 2M
        rope_kernel<<<pairs / 256, 256, 0, stream>>>(q, k, cos_t, sin_t);
    }
    // 3) causal flash attention (transposed orientation)
    {
        dim3 grid(CS / 256, CB * CH);
        flash_attn_kernel<<<grid, 256, 0, stream>>>(q, k, v, attn);
    }
    // 4) output projection: (4096 x 1024) @ (1024 x 1024) -> f32 d_out
    {
        dim3 grid(4096 / 128, 1024 / 128);
        gemm_bf16_wmma<false, 1><<<grid, 256, 0, stream>>>(
            attn, w_out, b_out, nullptr, (float*)d_out, CB * CS, CD, CD);
    }
}
